// Trans_62998580298318
// MI455X (gfx1250) — matrix operand order repaired and checked by reading, not compiled, not run on hardware
//
#include <hip/hip_runtime.h>

typedef __attribute__((ext_vector_type(2)))  float  v2f;
typedef __attribute__((ext_vector_type(4)))  float  v4f;
typedef __attribute__((ext_vector_type(8)))  float  v8f;
typedef __attribute__((ext_vector_type(8)))  __bf16 v8bf;
typedef __attribute__((ext_vector_type(16))) __bf16 v16bf;

static constexpr int B_N = 4;
static constexpr int C_N = 128;
static constexpr int L_N = 4096;
static constexpr size_t BCL = (size_t)B_N * C_N * L_N;   // 2,097,152
static constexpr size_t BLL = (size_t)B_N * L_N * L_N;   // 67,108,864

// ---------------- WMMA wrappers ----------------
__device__ __forceinline__ v8f wmma_f32(v2f a, v2f b, v8f c) {
  return __builtin_amdgcn_wmma_f32_16x16x4_f32(false, a, false, b, (short)0, c,
                                               false, false);
}
__device__ __forceinline__ v8f wmma_bf(v16bf a, v16bf b, v8f c) {
  return __builtin_amdgcn_wmma_f32_16x16x32_bf16(false, a, false, b, (short)0,
                                                 c, false, false);
}

// Concatenate two 8x bf16 register chunks into one 16x bf16 WMMA operand.
__device__ __forceinline__ v16bf cat8(v8bf a, v8bf b) {
  v16bf r;
#pragma unroll
  for (int i = 0; i < 8; ++i) { r[i] = a[i]; r[i + 8] = b[i]; }
  return r;
}

// Error-compensated bf16 split: x ~= (float)h + (float)l.
__device__ __forceinline__ void split_bf(float x, __bf16& h, __bf16& l) {
  h = (__bf16)x;
  l = (__bf16)(x - (float)h);
}

// ---------------------------------------------------------------------------
// Kernel 1: q/k/v = W @ x + bias (f32 WMMA 16x16x4, K=C=128), then emit the
// bf16 hi/lo operand forms needed downstream:
//   q -> qT  [b][l][c]                 (A-operand of  kcont = q^T k)
//   k -> kT  [b][l][c]                 (A-operand of  vcont = k^T v)
//        kB  [b][c/32][l][16 dwords]   (B-operand of  kcont, pair-packed)
//   v -> vB  pair-packed               (B-operand of  vcont)
//        vN  [b][c][l]                 (A-operand of  out = V Attn^T)
// Waves: p(3) * b(4) * otile(8) * ltile(256) = 24576.
// ---------------------------------------------------------------------------
__global__ __launch_bounds__(128) void qkv_kernel(
    const float* __restrict__ x,
    const float* __restrict__ Wq, const float* __restrict__ bq,
    const float* __restrict__ Wk, const float* __restrict__ bk,
    const float* __restrict__ Wv, const float* __restrict__ bv,
    __bf16* __restrict__ qT_h, __bf16* __restrict__ qT_l,
    __bf16* __restrict__ kT_h, __bf16* __restrict__ kT_l,
    __bf16* __restrict__ kB_h, __bf16* __restrict__ kB_l,
    __bf16* __restrict__ vB_h, __bf16* __restrict__ vB_l,
    __bf16* __restrict__ vN_h, __bf16* __restrict__ vN_l) {
  const int lane = threadIdx.x & 31;
  int gw = blockIdx.x * 4 + (threadIdx.x >> 5);
  const int ltile = gw & 255; gw >>= 8;
  const int otile = gw & 7;   gw >>= 3;
  const int b     = gw & 3;   gw >>= 2;
  const int p     = gw;                           // 0=q, 1=k, 2=v

  const float* W    = (p == 0) ? Wq : (p == 1) ? Wk : Wv;
  const float* bias = (p == 0) ? bq : (p == 1) ? bk : bv;

  const int n   = lane & 15;                      // N inside tile (pos l)
  const int kh  = lane >> 4;                      // f32-A K-pair selector
  const int off = (lane < 16) ? 0 : 8;            // C/D row offset
  const int o0 = otile * 16, l0 = ltile * 16;
  const float* xb = x + (size_t)b * C_N * L_N;

  v8f acc = {};
  for (int c0 = 0; c0 < C_N; c0 += 4) {
    const int c = c0 + 2 * kh;
    v2f a, bb;
    a.x  = W[(size_t)(o0 + n) * C_N + c];
    a.y  = W[(size_t)(o0 + n) * C_N + c + 1];
    bb.x = xb[(size_t)c * L_N + l0 + n];
    bb.y = xb[(size_t)(c + 1) * L_N + l0 + n];
    acc = wmma_f32(a, bb, acc);
  }

  // Lane holds channels o0+off .. o0+off+7 at position l0+n.
  v8bf h, l;
#pragma unroll
  for (int r = 0; r < 8; ++r) {
    const float val = acc[r] + bias[o0 + off + r];
    __bf16 hh, ll;
    split_bf(val, hh, ll);
    h[r] = hh; l[r] = ll;
  }

  const int    pos   = l0 + n;
  const size_t tIdx  = (size_t)b * L_N * C_N + (size_t)pos * C_N + o0 + off;
  const int    cb    = o0 >> 5;
  const size_t bIdx  = (size_t)b * C_N * L_N +
                       ((size_t)cb * L_N + pos) * 32 + (o0 & 31) + off;

  if (p == 0) {
    *(v8bf*)(qT_h + tIdx) = h;
    *(v8bf*)(qT_l + tIdx) = l;
  } else if (p == 1) {
    *(v8bf*)(kT_h + tIdx) = h;
    *(v8bf*)(kT_l + tIdx) = l;
    *(v8bf*)(kB_h + bIdx) = h;
    *(v8bf*)(kB_l + bIdx) = l;
  } else {
    *(v8bf*)(vB_h + bIdx) = h;
    *(v8bf*)(vB_l + bIdx) = l;
    const size_t nb = (size_t)b * C_N * L_N;
#pragma unroll
    for (int r = 0; r < 8; ++r) {
      vN_h[nb + (size_t)(o0 + off + r) * L_N + pos] = h[r];
      vN_l[nb + (size_t)(o0 + off + r) * L_N + pos] = l[r];
    }
  }
}

// A-operand load: two contiguous 16B chunks per lane (16-bit 16x32 layout).
__device__ __forceinline__ v16bf loadA(const __bf16* row, int kbase, int ab) {
  v8bf g0 = *(const v8bf*)(row + kbase + 2 * ab);
  v8bf g1 = *(const v8bf*)(row + kbase + 2 * ab + 8);
  return cat8(g0, g1);
}

// ---------------------------------------------------------------------------
// Kernel 2: kcont = q^T k, vcont = k^T v  (bf16x2 split WMMA, K=C=128).
// One wave computes both 16x16 tiles for (b, ltile, mtile); poscont = kcont.
// Waves: 4 * 256 * 256 = 262144.
// ---------------------------------------------------------------------------
__global__ __launch_bounds__(128) void gram_kernel(
    const __bf16* __restrict__ qT_h, const __bf16* __restrict__ qT_l,
    const __bf16* __restrict__ kT_h, const __bf16* __restrict__ kT_l,
    const __bf16* __restrict__ kB_h, const __bf16* __restrict__ kB_l,
    const __bf16* __restrict__ vB_h, const __bf16* __restrict__ vB_l,
    float* __restrict__ kcont, float* __restrict__ poscont,
    float* __restrict__ vcont) {
  const int lane = threadIdx.x & 31;
  int gw = blockIdx.x * 4 + (threadIdx.x >> 5);
  const int mtile = gw & 255; gw >>= 8;
  const int ltile = gw & 255; gw >>= 8;
  const int b     = gw;

  const int n  = lane & 15;
  const int ab = (lane < 16) ? 0 : 8;       // A chunk base (dword units)
  const int bb = (lane < 16) ? 0 : 16;      // B chunk base (bf16 units)
  const int l0 = ltile * 16, m0 = mtile * 16;

  const __bf16* qTrow_h = qT_h + (size_t)b * L_N * C_N + (size_t)(l0 + n) * C_N;
  const __bf16* qTrow_l = qT_l + (size_t)b * L_N * C_N + (size_t)(l0 + n) * C_N;
  const __bf16* kTrow_h = kT_h + (size_t)b * L_N * C_N + (size_t)(l0 + n) * C_N;
  const __bf16* kTrow_l = kT_l + (size_t)b * L_N * C_N + (size_t)(l0 + n) * C_N;
  const size_t  bbase   = (size_t)b * C_N * L_N;

  v8f accQK = {};
  v8f accKV = {};
#pragma unroll
  for (int cb = 0; cb < 4; ++cb) {
    const int c0 = cb * 32;
    const size_t bo = bbase + ((size_t)cb * L_N + m0 + n) * 32 + bb;

    v16bf Aqh = loadA(qTrow_h, c0, ab);
    v16bf Aql = loadA(qTrow_l, c0, ab);
    v16bf Akh = loadA(kTrow_h, c0, ab);
    v16bf Akl = loadA(kTrow_l, c0, ab);

    v16bf Bkh = cat8(*(const v8bf*)(kB_h + bo), *(const v8bf*)(kB_h + bo + 8));
    v16bf Bkl = cat8(*(const v8bf*)(kB_l + bo), *(const v8bf*)(kB_l + bo + 8));
    v16bf Bvh = cat8(*(const v8bf*)(vB_h + bo), *(const v8bf*)(vB_h + bo + 8));
    v16bf Bvl = cat8(*(const v8bf*)(vB_l + bo), *(const v8bf*)(vB_l + bo + 8));

    // 3-term compensated product: Ah*Bh + Ah*Bl + Al*Bh (f32 accumulate).
    accQK = wmma_bf(Aqh, Bkh, wmma_bf(Aqh, Bkl, wmma_bf(Aql, Bkh, accQK)));
    accKV = wmma_bf(Akh, Bvh, wmma_bf(Akh, Bvl, wmma_bf(Akl, Bvh, accKV)));
  }

  const size_t base = (size_t)b * L_N * L_N;
#pragma unroll
  for (int r = 0; r < 8; ++r) {
    const int lrow = r + ((lane < 16) ? 0 : 8);
    const size_t idx = base + (size_t)(l0 + lrow) * L_N + m0 + n;
    kcont[idx] = accQK[r];                                // re-read later
    __builtin_nontemporal_store(accQK[r], &poscont[idx]); // write-once
    __builtin_nontemporal_store(accKV[r], &vcont[idx]);   // write-once
  }
}

// ---------------------------------------------------------------------------
// Kernel 3: per-row softmax stats for energy = 2*kcont.
// ---------------------------------------------------------------------------
__global__ __launch_bounds__(256) void stats_kernel(
    const float* __restrict__ kcont, float* __restrict__ mx,
    float* __restrict__ rs) {
  const size_t row = blockIdx.x;
  const float* r = kcont + row * L_N;
  __shared__ float red[256];

  float m = -3.402823466e38f;
  for (int i = threadIdx.x; i < L_N; i += 256) m = fmaxf(m, 2.0f * r[i]);
  red[threadIdx.x] = m;
  __syncthreads();
  for (int s = 128; s > 0; s >>= 1) {
    if ((int)threadIdx.x < s)
      red[threadIdx.x] = fmaxf(red[threadIdx.x], red[threadIdx.x + s]);
    __syncthreads();
  }
  m = red[0];
  __syncthreads();

  float sum = 0.0f;
  for (int i = threadIdx.x; i < L_N; i += 256) sum += __expf(2.0f * r[i] - m);
  red[threadIdx.x] = sum;
  __syncthreads();
  for (int s = 128; s > 0; s >>= 1) {
    if ((int)threadIdx.x < s) red[threadIdx.x] += red[threadIdx.x + s];
    __syncthreads();
  }
  if (threadIdx.x == 0) {
    mx[row] = m;
    rs[row] = 1.0f / red[0];
  }
}

// ---------------------------------------------------------------------------
// Kernel 4: out[b,c,m] = sum_l v[b,c,l] * attn[b,m,l];
// attn built on the fly from kcont + row stats, split to bf16 hi/lo.
// Each wave: one m-tile, half of C (4 c-tiles), K = L = 4096 in 32-chunks.
// Waves: 4 * 256 * 2 = 2048.
// ---------------------------------------------------------------------------
__global__ __launch_bounds__(128) void out_kernel(
    const __bf16* __restrict__ vN_h, const __bf16* __restrict__ vN_l,
    const float* __restrict__ kcont, const float* __restrict__ mx,
    const float* __restrict__ rs, float* __restrict__ out) {
  const int lane = threadIdx.x & 31;
  int gw = blockIdx.x * 4 + (threadIdx.x >> 5);
  const int chalf = gw & 1;   gw >>= 1;
  const int mtile = gw & 255; gw >>= 8;
  const int b     = gw;

  const int n  = lane & 15;
  const int ab = (lane < 16) ? 0 : 8;       // A chunk base (dword units)
  const int bo = (lane < 16) ? 0 : 16;      // B-side l offset
  const int m0 = mtile * 16;
  const int c0 = chalf * 64;

  const size_t nb  = (size_t)b * C_N * L_N;
  const float* kc  = kcont + (size_t)b * L_N * L_N + (size_t)(m0 + n) * L_N;
  const float  mxv = mx[b * L_N + m0 + n];
  const float  riv = rs[b * L_N + m0 + n];

  v8f acc[4] = {{}, {}, {}, {}};
  for (int lb = 0; lb < L_N; lb += 32) {
    // Build attention B operand: 16 consecutive f32 -> exp -> bf16 hi/lo.
    const float* ap = kc + lb + bo;
    v4f q0 = *(const v4f*)(ap + 0);
    v4f q1 = *(const v4f*)(ap + 4);
    v4f q2 = *(const v4f*)(ap + 8);
    v4f q3 = *(const v4f*)(ap + 12);
    float a16[16];
#pragma unroll
    for (int i = 0; i < 4; ++i) {
      a16[i]      = q0[i];
      a16[i + 4]  = q1[i];
      a16[i + 8]  = q2[i];
      a16[i + 12] = q3[i];
    }
    v16bf Bh, Bl;
#pragma unroll
    for (int i = 0; i < 16; ++i) {
      const float attn = __expf(2.0f * a16[i] - mxv) * riv;
      __bf16 hh, ll;
      split_bf(attn, hh, ll);
      Bh[i] = hh; Bl[i] = ll;
    }
#pragma unroll
    for (int ct = 0; ct < 4; ++ct) {
      const __bf16* rowh = vN_h + nb + (size_t)(c0 + ct * 16 + n) * L_N;
      const __bf16* rowl = vN_l + nb + (size_t)(c0 + ct * 16 + n) * L_N;
      v16bf Ah = loadA(rowh, lb, ab);
      v16bf Al = loadA(rowl, lb, ab);
      acc[ct] = wmma_bf(Ah, Bh, wmma_bf(Ah, Bl, wmma_bf(Al, Bh, acc[ct])));
    }
  }

  float* ob = out + (size_t)b * C_N * L_N;
#pragma unroll
  for (int ct = 0; ct < 4; ++ct) {
#pragma unroll
    for (int r = 0; r < 8; ++r) {
      const int c = c0 + ct * 16 + r + ((lane < 16) ? 0 : 8);
      __builtin_nontemporal_store(acc[ct][r], &ob[(size_t)c * L_N + m0 + n]);
    }
  }
}

// ---------------------------------------------------------------------------
// d_out layout: out[BCL] | kcont[BLL] | poscont[BLL] | vcont[BLL].
// Workspace: 10 bf16 arrays of BCL (40 MB) + mx/rs (128 KB).
// ---------------------------------------------------------------------------
extern "C" void kernel_launch(void* const* d_in, const int* in_sizes, int n_in,
                              void* d_out, int out_size, void* d_ws,
                              size_t ws_size, hipStream_t stream) {
  const float* x  = (const float*)d_in[0];
  const float* Wq = (const float*)d_in[1];
  const float* bq = (const float*)d_in[2];
  const float* Wk = (const float*)d_in[3];
  const float* bk = (const float*)d_in[4];
  const float* Wv = (const float*)d_in[5];
  const float* bv = (const float*)d_in[6];

  float* out     = (float*)d_out;
  float* kcont   = out + BCL;
  float* poscont = kcont + BLL;
  float* vcont   = poscont + BLL;

  __bf16* wb   = (__bf16*)d_ws;
  __bf16* qT_h = wb + 0 * BCL;
  __bf16* qT_l = wb + 1 * BCL;
  __bf16* kT_h = wb + 2 * BCL;
  __bf16* kT_l = wb + 3 * BCL;
  __bf16* kB_h = wb + 4 * BCL;
  __bf16* kB_l = wb + 5 * BCL;
  __bf16* vB_h = wb + 6 * BCL;
  __bf16* vB_l = wb + 7 * BCL;
  __bf16* vN_h = wb + 8 * BCL;
  __bf16* vN_l = wb + 9 * BCL;
  float*  mx   = (float*)(wb + 10 * BCL);
  float*  rs   = mx + (size_t)B_N * L_N;

  qkv_kernel<<<6144, 128, 0, stream>>>(x, Wq, bq, Wk, bk, Wv, bv,
                                       qT_h, qT_l, kT_h, kT_l, kB_h, kB_l,
                                       vB_h, vB_l, vN_h, vN_l);
  gram_kernel<<<65536, 128, 0, stream>>>(qT_h, qT_l, kT_h, kT_l, kB_h, kB_l,
                                         vB_h, vB_l, kcont, poscont, vcont);
  stats_kernel<<<B_N * L_N, 256, 0, stream>>>(kcont, mx, rs);
  out_kernel<<<512, 128, 0, stream>>>(vN_h, vN_l, kcont, mx, rs, out);
}
